// MultiHeadSelfAttention_43568148250974
// MI455X (gfx1250) — matrix operand order repaired and checked by reading, not compiled, not run on hardware
//
#include <hip/hip_runtime.h>
#include <hip/hip_bf16.h>

// ---------------------------------------------------------------------------
// MultiHeadSelfAttention forward for MI455X (gfx1250, wave32, WMMA bf16 + TDM)
//   B=2, S=2048, D=1024, H=16, Dh=64
// ---------------------------------------------------------------------------

typedef __attribute__((ext_vector_type(16))) __bf16 v16bf;
typedef __attribute__((ext_vector_type(8)))  __bf16 v8bf;
typedef __attribute__((ext_vector_type(8)))  float  v8f;
typedef __attribute__((ext_vector_type(4)))  unsigned int v4u;
typedef __attribute__((ext_vector_type(8)))  int    v8i;
typedef __attribute__((ext_vector_type(4)))  int    v4i;

constexpr int Bsz = 2;
constexpr int Sq  = 2048;
constexpr int Dm  = 1024;
constexpr int Hh  = 16;
constexpr int Dh  = 64;
constexpr int KP  = 72;   // padded LDS row pitch in elements (128B row + 16B pad)

__device__ __forceinline__ v8bf ld8bf(const __bf16* p) {
  v4u u = *(const v4u*)p;
  return __builtin_bit_cast(v8bf, u);
}
__device__ __forceinline__ v16bf cat16(v8bf lo, v8bf hi) {
  return __builtin_shufflevector(lo, hi, 0,1,2,3,4,5,6,7,8,9,10,11,12,13,14,15);
}
__device__ __forceinline__ v16bf ld16bf(const __bf16* p) {  // 32 contiguous bytes
  return cat16(ld8bf(p), ld8bf(p + 8));
}
__device__ __forceinline__ __bf16 f2bf(float f) {           // round-to-nearest-even
  unsigned u = __builtin_bit_cast(unsigned, f);
  unsigned r = u + 0x7FFFu + ((u >> 16) & 1u);
  unsigned short h = (unsigned short)(r >> 16);
  return __builtin_bit_cast(__bf16, h);
}
__device__ __forceinline__ v8f wmma_bf16(v16bf a, v16bf b, v8f c) {
  return __builtin_amdgcn_wmma_f32_16x16x32_bf16(false, a, false, b, (short)0, c,
                                                 false, false);
}

// ---------------------------------------------------------------------------
// Tensor Data Mover: 2D tile (bf16) global -> LDS, padded 16B per 128B row.
// D# per CDNA5 ISA (group0 128b, group1 256b).
// This toolchain exposes the 6-arg builtin form:
//   (uint32x4 g0, int32x8 g1, int32x4 g2, int32x4 g3, int32x8 gx, i32 cpol)
// ---------------------------------------------------------------------------
#if defined(__has_builtin)
#if __has_builtin(__builtin_amdgcn_tensor_load_to_lds)
#define HAVE_TDM 1
#endif
#endif

#ifdef HAVE_TDM
__device__ __forceinline__ void tdm_load_2d(unsigned lds_addr, const void* gptr,
                                            unsigned tensor_d0, unsigned tensor_d1,
                                            unsigned tile_d0, unsigned tile_d1,
                                            unsigned stride0) {
  unsigned long long ga = (unsigned long long)gptr;
  v4u g0 = { 1u,                                   // count=1, user mode
             lds_addr,                             // LDS byte address
             (unsigned)ga,                         // global_addr[31:0]
             ((unsigned)(ga >> 32) & 0x01FFFFFFu) | (2u << 30) };  // addr[56:32]|type=2
  v8i g1;
  // data_size=2B(code1) | pad_enable | pad_interval=32dw(code4) | pad_amount=4dw(code3)
  g1.s0 = (int)((1u << 16) | (1u << 20) | (4u << 22) | (3u << 25));
  g1.s1 = (int)((tensor_d0 & 0xFFFFu) << 16);                   // dim0[15:0]
  g1.s2 = (int)((tensor_d0 >> 16) | ((tensor_d1 & 0xFFFFu) << 16));
  g1.s3 = (int)((tensor_d1 >> 16) | (tile_d0 << 16));
  g1.s4 = (int)(tile_d1 & 0xFFFFu);                              // tile_dim2 = 0
  g1.s5 = (int)stride0;                                          // dim0_stride lo
  g1.s6 = 0;
  g1.s7 = 0;
  v4i z4 = {0, 0, 0, 0};
  v8i z8 = {0, 0, 0, 0, 0, 0, 0, 0};
  __builtin_amdgcn_tensor_load_to_lds(g0, g1, z4, z4, z8, 0);
}
#define TDM_WAIT(n) __builtin_amdgcn_s_wait_tensorcnt((short)(n))
#else
// Fallback: wave-0 cooperative copy into the same padded layout.
__device__ __forceinline__ void tdm_load_2d(unsigned, const void*, unsigned, unsigned,
                                            unsigned, unsigned, unsigned) {}
#define TDM_WAIT(n) ((void)0)
#endif

__device__ __forceinline__ unsigned lds_off(const void* p) {
  return (unsigned)(unsigned long long)p;   // flat LDS addr: low 32 bits = LDS byte addr
}

#ifndef HAVE_TDM
__device__ __forceinline__ void wave_copy_tile(__bf16* dst, const __bf16* src,
                                               int srcPitch) {
  const int lane = threadIdx.x & 31;
  for (int r = lane; r < 64; r += 32) {
    const __bf16* s = src + (size_t)r * srcPitch;
    __bf16* d = dst + r * KP;
#pragma unroll
    for (int c = 0; c < 64; c += 8) *(v4u*)(d + c) = *(const v4u*)(s + c);
  }
}
#endif

// ---------------------------------------------------------------------------
// fp32 -> bf16 one-time conversion
// ---------------------------------------------------------------------------
__global__ void cvt_f32_bf16(const float* __restrict__ in, __bf16* __restrict__ out,
                             int n) {
  int i = blockIdx.x * blockDim.x + threadIdx.x;
  int stride = gridDim.x * blockDim.x;
  for (; i < n; i += stride) out[i] = f2bf(in[i]);
}

__global__ void zero_f32(float* __restrict__ p, int n) {
  int i = blockIdx.x * blockDim.x + threadIdx.x;
  if (i < n) p[i] = 0.0f;
}

// ---------------------------------------------------------------------------
// Kernel 1: QKV projection  Y = X @ W^T + b  (NT GEMM)
//   Wave tile 32(M) x 64(N): 8 WMMA per 32-wide K step, shared B operands.
// ---------------------------------------------------------------------------
__global__ void __launch_bounds__(256)
qkv_gemm(const __bf16* __restrict__ xb, const __bf16* __restrict__ wb,
         const float* __restrict__ bq, const float* __restrict__ bk,
         const float* __restrict__ bv,
         __bf16* __restrict__ qb, __bf16* __restrict__ kb,
         __bf16* __restrict__ vtb) {
  const int lane = threadIdx.x & 31;
  const int wave = threadIdx.x >> 5;
  const int which = blockIdx.z;                 // 0=Q 1=K 2=V
  const int m0 = blockIdx.x * 256 + wave * 32;  // rows of X (b*S+s)
  const int n0 = blockIdx.y * 64;               // one head's features
  const int l15 = lane & 15;
  const int hl  = (lane >> 4) & 1;

  const __bf16* W = wb + (size_t)which * Dm * Dm;
  const float*  bias = (which == 0) ? bq : (which == 1) ? bk : bv;
  __bf16*       dst  = (which == 0) ? qb : (which == 1) ? kb : vtb;

  const __bf16* arow0 = xb + (size_t)(m0 + l15) * Dm;
  const __bf16* arow1 = xb + (size_t)(m0 + 16 + l15) * Dm;
  const __bf16* wr[4];
#pragma unroll
  for (int t = 0; t < 4; ++t) wr[t] = W + (size_t)(n0 + t * 16 + l15) * Dm;

  v8f acc[2][4];
#pragma unroll
  for (int r = 0; r < 2; ++r)
#pragma unroll
    for (int t = 0; t < 4; ++t) { v8f z = {}; acc[r][t] = z; }

  for (int k0 = 0; k0 < Dm; k0 += 32) {
    const int ak = k0 + hl * 8;
    v16bf A0 = cat16(ld8bf(arow0 + ak), ld8bf(arow0 + ak + 16));
    v16bf A1 = cat16(ld8bf(arow1 + ak), ld8bf(arow1 + ak + 16));
    const int bk_ = k0 + hl * 8;
#pragma unroll
    for (int t = 0; t < 4; ++t) {
      v16bf Bt = cat16(ld8bf(wr[t] + bk_), ld8bf(wr[t] + bk_ + 16));
      acc[0][t] = wmma_bf16(A0, Bt, acc[0][t]);
      acc[1][t] = wmma_bf16(A1, Bt, acc[1][t]);
    }
  }

  const int head = n0 >> 6;
  float bn[4];
#pragma unroll
  for (int t = 0; t < 4; ++t) bn[t] = bias[n0 + t * 16 + l15];

#pragma unroll
  for (int r = 0; r < 2; ++r) {
#pragma unroll
    for (int t = 0; t < 4; ++t) {
      const int dh = t * 16 + l15;
#pragma unroll
      for (int i = 0; i < 8; ++i) {
        const int row = m0 + r * 16 + i + hl * 8;   // b*S + s
        const int bbi = row >> 11;
        const int s   = row & (Sq - 1);
        const __bf16 h16 = f2bf(acc[r][t][i] + bn[t]);
        if (which < 2) {
          dst[(((size_t)bbi * Hh + head) * Sq + s) * Dh + dh] = h16;
        } else {      // V stored transposed [B,H,Dh,S]
          dst[(((size_t)bbi * Hh + head) * Dh + dh) * Sq + s] = h16;
        }
      }
    }
  }
}

// ---------------------------------------------------------------------------
// Kernel 2: attention. Block = 4 waves sharing one (b,h); wave = 16 queries.
//   K / V^T tiles staged into LDS by TDM (double-buffered, wave-0 issued).
// ---------------------------------------------------------------------------
__global__ void __launch_bounds__(128)
attn_kernel(const __bf16* __restrict__ qb, const __bf16* __restrict__ kb,
            const __bf16* __restrict__ vtb, const float* __restrict__ mask,
            float* __restrict__ outctx, float* __restrict__ oscore) {
  __shared__ __bf16 kst[2][64 * KP];   // 2 x 9 KB
  __shared__ __bf16 vst[2][64 * KP];   // 2 x 9 KB
  __shared__ __bf16 pbuf[4][16 * 32];  // 4 KB

  const int lane = threadIdx.x & 31;
  const int wv   = threadIdx.x >> 5;
  const int blk  = blockIdx.x;                 // 1024 blocks
  const int qt   = (blk & 31) * 4 + wv;        // query tile 0..127
  const int h    = (blk >> 5) & (Hh - 1);
  const int b    = blk >> 9;
  const int l15  = lane & 15;
  const int hl   = (lane >> 4) & 1;

  const __bf16* qhead = qb  + ((size_t)b * Hh + h) * Sq * Dh;
  const __bf16* khead = kb  + ((size_t)b * Hh + h) * Sq * Dh;
  const __bf16* vhead = vtb + ((size_t)b * Hh + h) * Dh * Sq;
  const float*  mrow  = mask + (size_t)b * Sq;
  __bf16*       pb    = pbuf[wv];

  // Q as WMMA A operand, two 32-wide K chunks
  const __bf16* qrow = qhead + (size_t)(qt * 16 + l15) * Dh;
  const v16bf qa0 = ld16bf(qrow + hl * 16);
  const v16bf qa1 = ld16bf(qrow + 32 + hl * 16);

  const float scale = 0.125f;  // 1/sqrt(Dh)
  constexpr int NB = Sq / 64;  // 32 key blocks of 64

  float m_[8], l_[8];
#pragma unroll
  for (int i = 0; i < 8; ++i) { m_[i] = -1e30f; l_[i] = 0.0f; }

  // ---------------- Pass A: running (max, sumexp) ----------------
#ifdef HAVE_TDM
  if (wv == 0)
    tdm_load_2d(lds_off(kst[0]), khead, Dh, Sq, 64, 64, Dh);
#endif
  for (int kb64 = 0; kb64 < NB; ++kb64) {
    if (wv == 0) {
#ifdef HAVE_TDM
      if (kb64 + 1 < NB) {
        tdm_load_2d(lds_off(kst[(kb64 + 1) & 1]), khead + (size_t)(kb64 + 1) * 64 * Dh,
                    Dh, Sq, 64, 64, Dh);
        TDM_WAIT(1);
      } else {
        TDM_WAIT(0);
      }
#else
      wave_copy_tile(kst[kb64 & 1], khead + (size_t)kb64 * 64 * Dh, Dh);
#endif
    }
    __syncthreads();
    const __bf16* kt = kst[kb64 & 1];
#pragma unroll
    for (int j = 0; j < 4; ++j) {
      const __bf16* kr = kt + (j * 16 + l15) * KP;
      v16bf B0 = ld16bf(kr + hl * 16);
      v16bf B1 = ld16bf(kr + 32 + hl * 16);
      v8f cs = {};
      cs = wmma_bf16(qa0, B0, cs);
      cs = wmma_bf16(qa1, B1, cs);
      const float mk = mrow[kb64 * 64 + j * 16 + l15];
#pragma unroll
      for (int i = 0; i < 8; ++i) {
        const float s  = cs[i] * scale + mk;
        const float mn = fmaxf(m_[i], s);
        l_[i] = l_[i] * __expf(m_[i] - mn) + __expf(s - mn);
        m_[i] = mn;
      }
    }
    __syncthreads();
  }
  // merge (m,l) across the 16 lanes of each half (log-sum-exp merge)
#pragma unroll
  for (int off = 1; off < 16; off <<= 1) {
#pragma unroll
    for (int i = 0; i < 8; ++i) {
      const float om = __shfl_xor(m_[i], off, 32);
      const float ol = __shfl_xor(l_[i], off, 32);
      const float mn = fmaxf(m_[i], om);
      l_[i] = l_[i] * __expf(m_[i] - mn) + ol * __expf(om - mn);
      m_[i] = mn;
    }
  }
  float rl[8];
#pragma unroll
  for (int i = 0; i < 8; ++i) rl[i] = 1.0f / l_[i];

  v8f co[4];
#pragma unroll
  for (int t = 0; t < 4; ++t) { v8f z = {}; co[t] = z; }

  // ---------------- Pass B: normalize, column sums, O += P @ V ----------------
#ifdef HAVE_TDM
  if (wv == 0) {
    tdm_load_2d(lds_off(kst[0]), khead, Dh, Sq, 64, 64, Dh);
    tdm_load_2d(lds_off(vst[0]), vhead, Sq, Dh, 64, 64, Sq);
  }
#endif
  for (int kb64 = 0; kb64 < NB; ++kb64) {
    if (wv == 0) {
#ifdef HAVE_TDM
      if (kb64 + 1 < NB) {
        tdm_load_2d(lds_off(kst[(kb64 + 1) & 1]), khead + (size_t)(kb64 + 1) * 64 * Dh,
                    Dh, Sq, 64, 64, Dh);
        tdm_load_2d(lds_off(vst[(kb64 + 1) & 1]), vhead + (size_t)(kb64 + 1) * 64,
                    Sq, Dh, 64, 64, Sq);
        TDM_WAIT(2);
      } else {
        TDM_WAIT(0);
      }
#else
      wave_copy_tile(kst[kb64 & 1], khead + (size_t)kb64 * 64 * Dh, Dh);
      wave_copy_tile(vst[kb64 & 1], vhead + (size_t)kb64 * 64, Sq);
#endif
    }
    __syncthreads();
    const __bf16* kt = kst[kb64 & 1];
    const __bf16* vt = vst[kb64 & 1];
#pragma unroll
    for (int sub = 0; sub < 2; ++sub) {
#pragma unroll
      for (int t = 0; t < 2; ++t) {
        const int kl = sub * 32 + t * 16 + l15;         // key within 64-block
        const __bf16* kr = kt + kl * KP;
        v16bf B0 = ld16bf(kr + hl * 16);
        v16bf B1 = ld16bf(kr + 32 + hl * 16);
        v8f cs = {};
        cs = wmma_bf16(qa0, B0, cs);
        cs = wmma_bf16(qa1, B1, cs);
        const int key = kb64 * 64 + kl;
        const float mk = mrow[key];
        float csum = 0.0f;
#pragma unroll
        for (int i = 0; i < 8; ++i) {
          const float p = __expf(cs[i] * scale + mk - m_[i]) * rl[i];
          csum += p;
          pb[(i + hl * 8) * 32 + t * 16 + l15] = f2bf(p);  // row-major 16x32
        }
        csum += __shfl_xor(csum, 16, 32);
        if (lane < 16) atomicAdd(&oscore[(size_t)b * Sq + key], csum);
      }
      asm volatile("s_wait_dscnt 0" ::: "memory");
      const __bf16* pr = pb + l15 * 32 + hl * 16;        // P tile -> A layout
      v16bf pa = ld16bf(pr);
#pragma unroll
      for (int t = 0; t < 4; ++t) {
        const __bf16* vr = vt + (t * 16 + l15) * KP + sub * 32 + hl * 16;
        v16bf vb = ld16bf(vr);
        co[t] = wmma_bf16(pa, vb, co[t]);
      }
    }
    __syncthreads();
  }

  // ---- store context [B,S,D] fp32 ----
#pragma unroll
  for (int t = 0; t < 4; ++t) {
#pragma unroll
    for (int i = 0; i < 8; ++i) {
      const int row = qt * 16 + i + hl * 8;
      const int col = h * 64 + t * 16 + l15;
      outctx[((size_t)b * Sq + row) * Dm + col] = co[t][i];
    }
  }
}

// ---------------------------------------------------------------------------
extern "C" void kernel_launch(void* const* d_in, const int* in_sizes, int n_in,
                              void* d_out, int out_size, void* d_ws, size_t ws_size,
                              hipStream_t stream) {
  const float* x    = (const float*)d_in[0];
  const float* mask = (const float*)d_in[1];
  const float* Wq   = (const float*)d_in[2];
  const float* bq   = (const float*)d_in[3];
  const float* Wk   = (const float*)d_in[4];
  const float* bk   = (const float*)d_in[5];
  const float* Wv   = (const float*)d_in[6];
  const float* bv   = (const float*)d_in[7];

  float* ctx = (float*)d_out;                       // [B,S,D]
  float* osc = ctx + (size_t)Bsz * Sq * Dm;         // [B,S]

  // workspace: xb(8MB) | wb x3(6MB) | q(8MB) | k(8MB) | v^T(8MB) = 38MB
  __bf16* xb  = (__bf16*)d_ws;
  __bf16* wb  = xb  + (size_t)Bsz * Sq * Dm;
  __bf16* qbp = wb  + (size_t)3 * Dm * Dm;
  __bf16* kbp = qbp + (size_t)Bsz * Sq * Dm;
  __bf16* vtp = kbp + (size_t)Bsz * Sq * Dm;

  cvt_f32_bf16<<<2048, 256, 0, stream>>>(x,  xb,              Bsz * Sq * Dm);
  cvt_f32_bf16<<<1024, 256, 0, stream>>>(Wq, wb,              Dm * Dm);
  cvt_f32_bf16<<<1024, 256, 0, stream>>>(Wk, wb + Dm * Dm,    Dm * Dm);
  cvt_f32_bf16<<<1024, 256, 0, stream>>>(Wv, wb + 2 * Dm * Dm, Dm * Dm);
  zero_f32<<<(Bsz * Sq + 255) / 256, 256, 0, stream>>>(osc, Bsz * Sq);

  qkv_gemm<<<dim3(16, 16, 3), 256, 0, stream>>>(xb, wb, bq, bk, bv, qbp, kbp, vtp);
  attn_kernel<<<1024, 128, 0, stream>>>(qbp, kbp, vtp, mask, ctx, osc);
}
